// SparseUNet_64785286693639
// MI455X (gfx1250) — matrix operand order repaired and checked by reading, not compiled, not run on hardware
//
#include <hip/hip_runtime.h>
#include <hip/hip_bf16.h>

typedef _Float16 f16;
typedef _Float16 v16h __attribute__((ext_vector_type(16)));
typedef _Float16 v8h  __attribute__((ext_vector_type(8)));
typedef float    v8f  __attribute__((ext_vector_type(8)));

#define BN_EPS 1e-5f

// ---------------------------------------------------------------------------
// Weight staging: f32 [M][Cin][taps] -> f16 tap-major [M][taps*Cin]
//   dst[m][t*Cin + ci] = (f16) src[(m*Cin + ci)*taps + t]
// taps==1 degenerates to a plain cast (FC layers).
// ---------------------------------------------------------------------------
__global__ void k_wperm(const float* __restrict__ src, f16* __restrict__ dst,
                        int M, int Cin, int taps) {
  int KT = Cin * taps;
  int total = M * KT;
  int stride = gridDim.x * blockDim.x;
  for (int i = blockIdx.x * blockDim.x + threadIdx.x; i < total; i += stride) {
    int m = i / KT;
    int r = i - m * KT;
    int t = r / Cin;
    int ci = r - t * Cin;
    dst[i] = (f16)src[(size_t)(m * Cin + ci) * taps + t];
  }
}

// ---------------------------------------------------------------------------
// Implicit-GEMM conv via WMMA, tap-major K, f32 accumulate.
// Activations are voxel-major [voxel][Cin] (f16): B fragments are contiguous.
//   mode 0: 3x3x3 'SAME'  (taps=27, input dims == output dims)
//   mode 1: 2x2x2 s2 down (taps=8,  input dims == 2*output)
//   mode 2: pointwise GEMM (taps=1)
// One wave computes a (16*MB)(M) x 64(N) strip: per 32-K step, 2*MB A-loads +
// 8 B-loads feed 4*MB WMMAs (each B fragment reused MB times).
// Output f32 is channel-major [M][N] (feeds the BN stats reduction).
// Requires M%(16*MB)==0, N%64==0, Cin%32==0.
// ---------------------------------------------------------------------------
template <int MB>
__global__ void k_conv_wmma(const f16* __restrict__ Wt, const f16* __restrict__ act,
                            float* __restrict__ out,
                            int M, int Cin, int taps, int D, int H, int Wd,
                            int mode, int total_waves) {
  int wave = blockIdx.x * (blockDim.x >> 5) + (threadIdx.x >> 5);
  if (wave >= total_waves) return;          // whole-wave exit: EXEC all-ones for WMMA
  int lane = threadIdx.x & 31;
  int grp = lane >> 4;
  int lid = lane & 15;
  int N = D * H * Wd;
  int tiles_n = N >> 6;
  int mt = wave / tiles_n;
  int nt = wave - mt * tiles_n;
  int K = Cin * taps;

  // the 4 voxel columns this lane owns (constant over K loop)
  int nv[4], od[4], oh[4], ow[4];
#pragma unroll
  for (int j = 0; j < 4; ++j) {
    int n = (nt << 6) + (j << 4) + lid;
    nv[j] = n;
    int d = n / (H * Wd);
    int r = n - d * H * Wd;
    int h = r / Wd;
    od[j] = d; oh[j] = h; ow[j] = r - h * Wd;
  }

  // A row pointers: one per M sub-tile
  const f16* arowbase[MB];
#pragma unroll
  for (int mb = 0; mb < MB; ++mb) {
    int mrow = mt * (MB << 4) + (mb << 4) + lid;
    arowbase[mb] = Wt + (size_t)mrow * K + grp * 8;
  }

  v8f c[MB][4];
#pragma unroll
  for (int mb = 0; mb < MB; ++mb)
#pragma unroll
    for (int j = 0; j < 4; ++j) c[mb][j] = (v8f){};

  for (int t = 0; t < taps; ++t) {
    // input voxel index + bounds: once per tap, per owned voxel column
    int sj[4], ib[4];
    if (mode == 0) {
      int dd = t / 9; int rr = t - dd * 9; int dh = rr / 3; int dw = rr - dh * 3;
#pragma unroll
      for (int j = 0; j < 4; ++j) {
        int id = od[j] + dd - 1, ih = oh[j] + dh - 1, iw = ow[j] + dw - 1;
        ib[j] = ((unsigned)id < (unsigned)D) & ((unsigned)ih < (unsigned)H) &
                ((unsigned)iw < (unsigned)Wd);
        sj[j] = (id * H + ih) * Wd + iw;
      }
    } else if (mode == 1) {
#pragma unroll
      for (int j = 0; j < 4; ++j) {
        int id = (od[j] << 1) + (t >> 2);
        int ih = (oh[j] << 1) + ((t >> 1) & 1);
        int iw = (ow[j] << 1) + (t & 1);
        ib[j] = 1;
        sj[j] = (id * (H << 1) + ih) * (Wd << 1) + iw;
      }
    } else {
#pragma unroll
      for (int j = 0; j < 4; ++j) { ib[j] = 1; sj[j] = nv[j]; }
    }

    size_t toff = (size_t)t * Cin;
    for (int cb = 0; cb < Cin; cb += 32) {
      // A fragments: two aligned b128 loads each; K = (e/8)*16 + grp*8 + e%8
      v16h a[MB];
#pragma unroll
      for (int mb = 0; mb < MB; ++mb) {
        const f16* arow = arowbase[mb] + toff + cb;
        v8h alo = *reinterpret_cast<const v8h*>(arow);
        v8h ahi = *reinterpret_cast<const v8h*>(arow + 16);
#pragma unroll
        for (int e = 0; e < 8; ++e) { a[mb][e] = alo[e]; a[mb][e + 8] = ahi[e]; }
      }

#pragma unroll
      for (int j = 0; j < 4; ++j) {
        // B fragment: K = cb + (e/8)*16 + grp*8 + e%8 -> two 8-channel runs at voxel sj[j]
        v16h b = {};
        if (ib[j]) {
          const f16* bp = act + (size_t)sj[j] * Cin + cb + (grp << 3);
          v8h blo = *reinterpret_cast<const v8h*>(bp);
          v8h bhi = *reinterpret_cast<const v8h*>(bp + 16);
#pragma unroll
          for (int e = 0; e < 8; ++e) { b[e] = blo[e]; b[e + 8] = bhi[e]; }
        }
#pragma unroll
        for (int mb = 0; mb < MB; ++mb)
          c[mb][j] = __builtin_amdgcn_wmma_f32_16x16x32_f16(false, a[mb], false, b,
                                                            (short)0, c[mb][j],
                                                            false, false);
      }
    }
  }

  // C/D layout: N = lane%16, M = mt*16*MB + mb*16 + grp*8 + r
#pragma unroll
  for (int mb = 0; mb < MB; ++mb) {
#pragma unroll
    for (int j = 0; j < 4; ++j) {
#pragma unroll
      for (int r = 0; r < 8; ++r) {
        int row = mt * (MB << 4) + (mb << 4) + (grp << 3) + r;
        out[(size_t)row * N + nv[j]] = c[mb][j][r];
      }
    }
  }
}

// ---------------------------------------------------------------------------
// Per-channel BatchNorm stats (train mode, biased var) over N voxels.
// Input: channel-major f32 [C][N] (GEMM output).
// ---------------------------------------------------------------------------
__global__ void k_stats(const float* __restrict__ x, float* __restrict__ mean,
                        float* __restrict__ rsig, int N) {
  __shared__ float ssum[256];
  __shared__ float ssq[256];
  int c = blockIdx.x;
  const float* p = x + (size_t)c * N;
  float s = 0.f, q = 0.f;
  for (int i = threadIdx.x; i < N; i += blockDim.x) { float v = p[i]; s += v; q += v * v; }
  ssum[threadIdx.x] = s; ssq[threadIdx.x] = q;
  __syncthreads();
  for (int st = 128; st > 0; st >>= 1) {
    if ((int)threadIdx.x < st) {
      ssum[threadIdx.x] += ssum[threadIdx.x + st];
      ssq[threadIdx.x]  += ssq[threadIdx.x + st];
    }
    __syncthreads();
  }
  if (threadIdx.x == 0) {
    float m = ssum[0] / (float)N;
    float v = ssq[0] / (float)N - m * m;
    mean[c] = m;
    rsig[c] = rsqrtf(v + BN_EPS);
  }
}

// ---------------------------------------------------------------------------
// BN normalize + ReLU + cast + transpose: f32 [C][N] -> f16 [N][C].
// 32(ch) x 64(vox) LDS tile; coalesced loads, packed v8h coalesced stores.
// Requires C%32==0, N%64==0.
// ---------------------------------------------------------------------------
__global__ void k_bn_relu_tr(const float* __restrict__ x, const float* __restrict__ mean,
                             const float* __restrict__ rsig, f16* __restrict__ y,
                             int C, int N) {
  __shared__ float tile[32][65];
  int tilesN = N >> 6;
  int tc = blockIdx.x / tilesN;
  int tn = blockIdx.x - tc * tilesN;
  int c0 = tc << 5;
  int n0 = tn << 6;

  int vtid = threadIdx.x & 63;
  int cbeg = threadIdx.x >> 6;
#pragma unroll
  for (int s = 0; s < 8; ++s) {
    int cc = cbeg + (s << 2);
    float v = x[(size_t)(c0 + cc) * N + n0 + vtid];
    v = (v - mean[c0 + cc]) * rsig[c0 + cc];
    tile[cc][vtid] = v > 0.f ? v : 0.f;
  }
  __syncthreads();

  int vv = threadIdx.x >> 2;
  int cseg = (threadIdx.x & 3) << 3;
  v8h outv;
#pragma unroll
  for (int e = 0; e < 8; ++e) outv[e] = (f16)tile[cseg + e][vv];
  *reinterpret_cast<v8h*>(y + (size_t)(n0 + vv) * C + c0 + cseg) = outv;
}

// ---------------------------------------------------------------------------
// in_conv: 1x1x1, 4 -> 32 channels (K=4 too small for WMMA; trivial FLOPs)
// x is [4][N] (reference layout); output channel-major f32 [32][N].
// ---------------------------------------------------------------------------
__global__ void k_inconv(const float* __restrict__ x, const float* __restrict__ w,
                         float* __restrict__ out, int N) {
  int total = 32 * N;
  int stride = gridDim.x * blockDim.x;
  for (int i = blockIdx.x * blockDim.x + threadIdx.x; i < total; i += stride) {
    int o = i / N;
    int n = i - o * N;
    float acc = 0.f;
#pragma unroll
    for (int c = 0; c < 4; ++c) acc += x[(size_t)c * N + n] * w[o * 4 + c];
    out[i] = acc;
  }
}

// ---------------------------------------------------------------------------
// Transposed conv 2x2x2 stride 2 (fractionally-strided; spatially flipped tap).
// Input f16 [Nin][Cin]; writes f16 into columns [0,Cout) of concat buffer
// [Nout][Ctot]. <2% of total FLOPs -> plain VALU kernel, coalesced stores.
// ---------------------------------------------------------------------------
__global__ void k_upconv(const f16* __restrict__ in, const float* __restrict__ w,
                         f16* __restrict__ out, int Cin, int Cout, int Ctot,
                         int Di, int Hi, int Wi) {
  int Ho = Hi << 1, Wo = Wi << 1;
  int Nov = (Di * Hi * Wi) << 3;
  int total = Nov * Cout;
  int stride = gridDim.x * blockDim.x;
  for (int i = blockIdx.x * blockDim.x + threadIdx.x; i < total; i += stride) {
    int n = i / Cout;
    int o = i - n * Cout;
    int od = n / (Ho * Wo);
    int r = n - od * Ho * Wo;
    int oh = r / Wo;
    int ow = r - oh * Wo;
    int id = od >> 1, ih = oh >> 1, iw = ow >> 1;
    int tap = ((1 - (od & 1)) << 2) | ((1 - (oh & 1)) << 1) | (1 - (ow & 1));
    const float* wp = w + (size_t)o * Cin * 8 + tap;
    const f16* ip = in + (size_t)((id * Hi + ih) * Wi + iw) * Cin;
    float acc = 0.f;
    for (int ci = 0; ci < Cin; ++ci)
      acc += (float)ip[ci] * wp[(size_t)ci * 8];
    out[(size_t)n * Ctot + o] = (f16)acc;
  }
}

// append skip [N][Csk] into columns [Co, Co+Csk) of concat buffer [N][Ctot]
__global__ void k_catcopy(const f16* __restrict__ src, f16* __restrict__ dst,
                          int Csk, int Co, int Ctot, int N) {
  int total = N * Csk;
  int stride = gridDim.x * blockDim.x;
  for (int i = blockIdx.x * blockDim.x + threadIdx.x; i < total; i += stride) {
    int n = i / Csk;
    int c = i - n * Csk;
    dst[(size_t)n * Ctot + Co + c] = src[i];
  }
}

// final bias add + [M=16][N] -> [N][16] transpose into d_out (f32)
__global__ void k_bias_tr(const float* __restrict__ tmp, const float* __restrict__ bias,
                          float* __restrict__ out, int N) {
  int total = 16 * N;
  int stride = gridDim.x * blockDim.x;
  for (int i = blockIdx.x * blockDim.x + threadIdx.x; i < total; i += stride) {
    int m = i / N;
    int n = i - m * N;
    out[(size_t)n * 16 + m] = tmp[i] + bias[m];
  }
}

// ---------------------------------------------------------------------------
// Host orchestration
// ---------------------------------------------------------------------------
static inline int cdiv_i(int a, int b) { return (a + b - 1) / b; }

extern "C" void kernel_launch(void* const* d_in, const int* in_sizes, int n_in,
                              void* d_out, int out_size, void* d_ws, size_t ws_size,
                              hipStream_t stream) {
  (void)in_sizes; (void)n_in; (void)out_size; (void)ws_size;

  const float* x    = (const float*)d_in[0];
  const float* in_w = (const float*)d_in[1];
  const float* enc_down[4] = {(const float*)d_in[2],  (const float*)d_in[3],
                              (const float*)d_in[4],  (const float*)d_in[5]};
  const float* enc_blk[4]  = {(const float*)d_in[6],  (const float*)d_in[7],
                              (const float*)d_in[8],  (const float*)d_in[9]};
  const float* dec_up[4]   = {(const float*)d_in[10], (const float*)d_in[11],
                              (const float*)d_in[12], (const float*)d_in[13]};
  const float* dec_blk[4]  = {(const float*)d_in[14], (const float*)d_in[15],
                              (const float*)d_in[16], (const float*)d_in[17]};
  const float* fc1  = (const float*)d_in[18];
  const float* fc2  = (const float*)d_in[19];
  const float* fc2b = (const float*)d_in[20];

  const int N0 = 64 * 64 * 64, N1 = 32 * 32 * 32, N2 = 16 * 16 * 16, N3 = 512, N4 = 64;

  // bump allocator over d_ws
  char* wsb = (char*)d_ws;
  size_t off = 0;
  auto alloc = [&](size_t bytes) -> char* {
    size_t p = (off + 255) & ~(size_t)255;
    off = p + bytes;
    return wsb + p;
  };
  f16*   wf16 = (f16*)  alloc((size_t)512 * 512 * 27 * 2);  // max weight (enc_blk[3])
  f16*   s0   = (f16*)  alloc((size_t)32  * N0 * 2);        // skips, voxel-major [N][C]
  f16*   s1   = (f16*)  alloc((size_t)64  * N1 * 2);
  f16*   s2   = (f16*)  alloc((size_t)128 * N2 * 2);
  f16*   s3   = (f16*)  alloc((size_t)256 * N3 * 2);
  f16*   bott = (f16*)  alloc((size_t)512 * N4 * 2);
  f16*   actA = (f16*)  alloc((size_t)160 * N0 * 2);        // concat buffer / ping
  f16*   actB = (f16*)  alloc((size_t)128 * N0 * 2);        // pong
  float* cf32 = (float*)alloc((size_t)128 * N0 * 4);        // conv f32 output [C][N]
  float* mean = (float*)alloc(512 * 4);
  float* rsig = (float*)alloc(512 * 4);

  auto conv = [&](const float* wsrc, const f16* actin, int M, int Cin, int taps,
                  int D, int H, int W, int mode, float* outf) {
    int wcount = M * Cin * taps;
    k_wperm<<<cdiv_i(wcount, 256), 256, 0, stream>>>(wsrc, wf16, M, Cin, taps);
    int N = D * H * W;
    if (M % 32 == 0) {
      int waves = (M / 32) * (N / 64);
      k_conv_wmma<2><<<cdiv_i(waves, 8), 256, 0, stream>>>(wf16, actin, outf,
                                                           M, Cin, taps, D, H, W,
                                                           mode, waves);
    } else {
      int waves = (M / 16) * (N / 64);
      k_conv_wmma<1><<<cdiv_i(waves, 8), 256, 0, stream>>>(wf16, actin, outf,
                                                           M, Cin, taps, D, H, W,
                                                           mode, waves);
    }
  };
  auto bn = [&](const float* src, f16* dst, int C, int N) {
    k_stats<<<C, 256, 0, stream>>>(src, mean, rsig, N);
    k_bn_relu_tr<<<(C / 32) * (N / 64), 256, 0, stream>>>(src, mean, rsig, dst, C, N);
  };

  // ---- in_conv: 4 -> 32 @ 64^3, BN + ReLU -> s0 [N0][32] (enc_outs[0])
  k_inconv<<<cdiv_i(32 * N0, 256), 256, 0, stream>>>(x, in_w, cf32, N0);
  bn(cf32, s0, 32, N0);

  // ---- encoder
  const int encC[5] = {32, 64, 128, 256, 512};
  const int Ns[5]   = {N0, N1, N2, N3, N4};
  const int Ds[5]   = {64, 32, 16, 8, 4};
  f16* skips[5] = {s0, s1, s2, s3, bott};
  for (int l = 0; l < 4; ++l) {
    int Ci = encC[l], Co = encC[l + 1];
    int Dn = Ds[l + 1], Nn = Ns[l + 1];
    // stride-2 2^3 down conv + BN + ReLU -> actA [Nn][Co]
    conv(enc_down[l], skips[l], Co, Ci, 8, Dn, Dn, Dn, 1, cf32);
    bn(cf32, actA, Co, Nn);
    // 3^3 SAME conv block + BN + ReLU -> skip[l+1] [Nn][Co]
    conv(enc_blk[l], actA, Co, Co, 27, Dn, Dn, Dn, 0, cf32);
    bn(cf32, skips[l + 1], Co, Nn);
  }

  // ---- decoder
  const int decC[5] = {512, 256, 128, 128, 128};
  f16* cur = bott;
  for (int l = 0; l < 4; ++l) {
    int Ci = decC[l], Co = decC[l + 1];
    int Csk = encC[3 - l];              // enc_outs[-(l+2)] channels
    int Din = Ds[4 - l];
    int Dout = Ds[3 - l];
    int Nout = Ns[3 - l];
    int Ctot = Co + Csk;
    // up conv -> columns [0,Co) of concat buffer actA [Nout][Ctot]
    k_upconv<<<cdiv_i(Nout * Co, 256), 256, 0, stream>>>(cur, dec_up[l], actA,
                                                         Ci, Co, Ctot, Din, Din, Din);
    // append skip -> columns [Co, Ctot)
    k_catcopy<<<cdiv_i(Nout * Csk, 256), 256, 0, stream>>>(skips[3 - l], actA,
                                                           Csk, Co, Ctot, Nout);
    // 3^3 SAME conv block + BN + ReLU -> actB [Nout][Co]
    conv(dec_blk[l], actA, Co, Ctot, 27, Dout, Dout, Dout, 0, cf32);
    bn(cf32, actB, Co, Nout);
    cur = actB;
  }

  // ---- per-voxel FC head: fc1 (128x128 GEMM) + BN1d + ReLU, then fc2 + bias
  conv(fc1, actB, 128, 128, 1, 64, 64, 64, 2, cf32);
  bn(cf32, actA, 128, N0);
  conv(fc2, actA, 16, 128, 1, 64, 64, 64, 2, cf32);
  k_bias_tr<<<cdiv_i(16 * N0, 256), 256, 0, stream>>>(cf32, fc2b, (float*)d_out, N0);
}
